// SuperGATNet_55155970015563
// MI455X (gfx1250) — matrix operand order repaired and checked by reading, not compiled, not run on hardware
//
#include <hip/hip_runtime.h>

typedef __bf16 bf16_t;
typedef __attribute__((ext_vector_type(16))) __bf16 v16bf;
typedef __attribute__((ext_vector_type(8)))  float  v8f;

#define NN    50000
#define EE    800000
#define HIDN  128
#define NCLSN 10
#define NGRP  512
#define NCONV 3
#define E2    (EE + NN)

// ---------------------------------------------------------------------------
// Fold eval-mode BatchNorm (mean=0,var=1) into the following weight matrix:
//   (h*s + b) @ W = h @ (s ⊙ W) + (b @ W)        s[f] = g[f]/sqrt(1+eps)
// Emits the effective weight TRANSPOSED in bf16:  Wt[o][f] = s[f]*W[f][o]
// so GEMM B-fragments become contiguous K-runs. One block, 128 threads.
// ---------------------------------------------------------------------------
__global__ void prep_weights(const float* __restrict__ W, const float* __restrict__ bng,
                             const float* __restrict__ bnb, const float* __restrict__ extra,
                             bf16_t* __restrict__ Wt, float* __restrict__ ebias, int has_extra)
{
    int o = threadIdx.x;                       // output column 0..127
    float inv = 1.0f / sqrtf(1.0f + 1e-5f);
    float acc = has_extra ? extra[o] : 0.0f;
    for (int f = 0; f < 128; ++f) {
        float w = W[f * 128 + o];
        Wt[o * 128 + f] = (bf16_t)(w * (bng[f] * inv));   // transposed
        acc += bnb[f] * w;
    }
    ebias[o] = acc;
}

// ---------------------------------------------------------------------------
// bf16 WMMA GEMM:  out[nrows x 128] = f(A[nrows x 128]) @ W[128 x 128] + ebias
// (W given transposed: Wt[n][k]).  f = identity or ReLU (relu_in).
// Block = 256 threads = 8 waves, 128x128 tile, K=128 staged once in LDS
// (padded stride 136 bf16 = 272B -> conflict-free b128 fragment loads).
// Each wave: 32M x 64N = 8 v8f accumulators, 4 K-steps of v_wmma 16x16x32 bf16.
// ---------------------------------------------------------------------------
__global__ __launch_bounds__(256)
void gemm_bf16_128(const float* __restrict__ A, const bf16_t* __restrict__ Wt,
                   const float* __restrict__ ebias, float* __restrict__ out,
                   int nrows, int relu_in)
{
    __shared__ __align__(16) bf16_t As[128 * 136];   // row-major  [m][k]
    __shared__ __align__(16) bf16_t Bt[128 * 136];   // transposed [n][k]
    const int tid = threadIdx.x;
    const int m0  = blockIdx.x * 128;

    // stage A tile: f32 global -> bf16 LDS (optional ReLU), 16B LDS stores
    for (int u = tid; u < 128 * 16; u += 256) {
        int r = u >> 4;
        int c = (u & 15) << 3;                        // 8 floats per unit
        int row = m0 + r;
        float4 f0, f1;
        if (row < nrows) {
            f0 = *(const float4*)(A + (size_t)row * 128 + c);
            f1 = *(const float4*)(A + (size_t)row * 128 + c + 4);
        } else {
            f0.x = f0.y = f0.z = f0.w = 0.f;
            f1.x = f1.y = f1.z = f1.w = 0.f;
        }
        if (relu_in) {
            f0.x = fmaxf(f0.x, 0.f); f0.y = fmaxf(f0.y, 0.f);
            f0.z = fmaxf(f0.z, 0.f); f0.w = fmaxf(f0.w, 0.f);
            f1.x = fmaxf(f1.x, 0.f); f1.y = fmaxf(f1.y, 0.f);
            f1.z = fmaxf(f1.z, 0.f); f1.w = fmaxf(f1.w, 0.f);
        }
        union { bf16_t h[8]; uint4 q; } pk;
        pk.h[0] = (bf16_t)f0.x; pk.h[1] = (bf16_t)f0.y;
        pk.h[2] = (bf16_t)f0.z; pk.h[3] = (bf16_t)f0.w;
        pk.h[4] = (bf16_t)f1.x; pk.h[5] = (bf16_t)f1.y;
        pk.h[6] = (bf16_t)f1.z; pk.h[7] = (bf16_t)f1.w;
        *(uint4*)(As + r * 136 + c) = pk.q;
    }
    // stage Bt tile: bf16 global (already transposed) -> LDS, 16B vectors
    for (int u = tid; u < 128 * 16; u += 256) {
        int r = u >> 4;
        int c = (u & 15) << 3;
        uint4 d = *(const uint4*)(Wt + r * 128 + c);
        *(uint4*)(Bt + r * 136 + c) = d;
    }
    __syncthreads();

    const int wave = tid >> 5, lane = tid & 31;
    const int half = lane >> 4, ml = lane & 15;
    const int wr = wave & 3, wc = wave >> 2;     // 4x2 wave grid
    const int mbase = wr * 32, nbase = wc * 64;

    v8f acc[2][4];
    #pragma unroll
    for (int nt = 0; nt < 4; ++nt) {
        float eb = ebias[nbase + nt * 16 + ml];  // C layout: col = n0 + (lane&15)
        #pragma unroll
        for (int mt = 0; mt < 2; ++mt)
            #pragma unroll
            for (int r = 0; r < 8; ++r) acc[mt][nt][r] = eb;
    }

    #pragma unroll
    for (int kk = 0; kk < 4; ++kk) {
        const int k0 = kk * 32;
        // A fragment (16-bit 16x32 ISA layout): two contiguous 8-elem K runs
        //   elems 0..7  : K = k0 + half*8 + 0..7
        //   elems 8..15 : K = k0 + 16 + half*8 + 0..7
        v16bf afr[2];
        #pragma unroll
        for (int mt = 0; mt < 2; ++mt) {
            const bf16_t* Ap = As + (mbase + mt * 16 + ml) * 136;
            union { v16bf v; uint4 q[2]; } fa;
            fa.q[0] = *(const uint4*)(Ap + k0 + half * 8);
            fa.q[1] = *(const uint4*)(Ap + k0 + 16 + half * 8);
            afr[mt] = fa.v;
        }
        // B fragment (32x16): lane = column n, same K order as A
        //   elems 0..7  : K = k0 + half*8 + 0..7
        //   elems 8..15 : K = k0 + 16 + half*8 + 0..7
        v16bf bfr[4];
        #pragma unroll
        for (int nt = 0; nt < 4; ++nt) {
            const bf16_t* Bp = Bt + (nbase + nt * 16 + ml) * 136 + k0 + half * 8;
            union { v16bf v; uint4 q[2]; } fb;
            fb.q[0] = *(const uint4*)(Bp);
            fb.q[1] = *(const uint4*)(Bp + 16);
            bfr[nt] = fb.v;
        }
        #pragma unroll
        for (int mt = 0; mt < 2; ++mt)
            #pragma unroll
            for (int nt = 0; nt < 4; ++nt)
                acc[mt][nt] = __builtin_amdgcn_wmma_f32_16x16x32_bf16(
                    false, afr[mt], false, bfr[nt], (short)0, acc[mt][nt], false, false);
    }

    #pragma unroll
    for (int mt = 0; mt < 2; ++mt)
        #pragma unroll
        for (int r = 0; r < 8; ++r) {
            int row = m0 + mbase + mt * 16 + half * 8 + r;
            if (row < nrows) {
                #pragma unroll
                for (int nt = 0; nt < 4; ++nt)
                    out[(size_t)row * 128 + nbase + nt * 16 + ml] = acc[mt][nt][r];
            }
        }
}

// ---------------------------------------------------------------------------
// Edge-phase kernels (wave32: one wave per edge, lane l owns channel 32h+l)
// ---------------------------------------------------------------------------
__device__ inline void atomicMaxF(float* addr, float v) {
    if (v >= 0.f) atomicMax((int*)addr, __float_as_int(v));
    else          atomicMin((unsigned int*)addr, __float_as_uint(v));
}

__global__ __launch_bounds__(256)
void layer_init(float* __restrict__ out, const float* __restrict__ bgat,
                float* __restrict__ mmax, float* __restrict__ ssum)
{
    int i = blockIdx.x * 256 + threadIdx.x;
    if (i < NN * 128) out[i] = bgat[i & 127];            // bias-initialized accumulator
    if (i < NN * 4) { mmax[i] = -3.402823466e+38f; ssum[i] = 0.f; }
}

__global__ __launch_bounds__(256)
void edge_alpha(const float* __restrict__ hh, const int* __restrict__ src,
                const int* __restrict__ dst, const float* __restrict__ attl,
                const float* __restrict__ attr, float* __restrict__ alpha,
                float* __restrict__ mmax)
{
    int e    = (blockIdx.x * 256 + threadIdx.x) >> 5;
    int lane = threadIdx.x & 31;
    if (e >= E2) return;
    int s = (e < EE) ? src[e] : (e - EE);
    int d = (e < EE) ? dst[e] : (e - EE);
    float v[12];
    #pragma unroll
    for (int h = 0; h < 4; ++h) {
        int c = h * 32 + lane;
        float hj = hh[(size_t)s * 128 + c];
        float hi = hh[(size_t)d * 128 + c];
        v[h]     = hi * hj;                  // MX logit
        v[4 + h] = hj * attl[h * 32 + lane]; // h_j . att_l
        v[8 + h] = hi * attr[h * 32 + lane]; // h_i . att_r
    }
    #pragma unroll
    for (int off = 16; off > 0; off >>= 1)
        #pragma unroll
        for (int j = 0; j < 12; ++j)
            v[j] += __shfl_xor(v[j], off, 32);

    float a0, a1, a2, a3;
    {
        float t;
        t = (v[4] + v[8])  * (1.f / (1.f + __expf(-v[0]))); a0 = (t > 0.f) ? t : 0.2f * t;
        t = (v[5] + v[9])  * (1.f / (1.f + __expf(-v[1]))); a1 = (t > 0.f) ? t : 0.2f * t;
        t = (v[6] + v[10]) * (1.f / (1.f + __expf(-v[2]))); a2 = (t > 0.f) ? t : 0.2f * t;
        t = (v[7] + v[11]) * (1.f / (1.f + __expf(-v[3]))); a3 = (t > 0.f) ? t : 0.2f * t;
    }
    if (lane < 4) {
        float a = (lane == 0) ? a0 : ((lane == 1) ? a1 : ((lane == 2) ? a2 : a3));
        alpha[(size_t)e * 4 + lane] = a;
        atomicMaxF(&mmax[d * 4 + lane], a);
    }
}

__global__ __launch_bounds__(256)
void edge_exp(const int* __restrict__ dst, const float* __restrict__ mmax,
              float* __restrict__ alpha, float* __restrict__ ssum)
{
    int i = blockIdx.x * 256 + threadIdx.x;
    if (i >= E2 * 4) return;
    int e = i >> 2, h = i & 3;
    int d = (e < EE) ? dst[e] : (e - EE);
    float ex = __expf(alpha[i] - mmax[d * 4 + h]);
    alpha[i] = ex;
    atomicAdd(&ssum[d * 4 + h], ex);
}

__global__ __launch_bounds__(256)
void edge_scatter(const float* __restrict__ hh, const int* __restrict__ src,
                  const int* __restrict__ dst, const float* __restrict__ alpha,
                  const float* __restrict__ ssum, float* __restrict__ out)
{
    int e    = (blockIdx.x * 256 + threadIdx.x) >> 5;
    int lane = threadIdx.x & 31;
    if (e >= E2) return;
    int s = (e < EE) ? src[e] : (e - EE);
    int d = (e < EE) ? dst[e] : (e - EE);
    #pragma unroll
    for (int h = 0; h < 4; ++h) {
        float w = alpha[(size_t)e * 4 + h] / (ssum[d * 4 + h] + 1e-16f);
        int c = h * 32 + lane;
        atomicAdd(&out[(size_t)d * 128 + c], hh[(size_t)s * 128 + c] * w);
    }
}

// ---------------------------------------------------------------------------
// Pooling + classifier head
// ---------------------------------------------------------------------------
__global__ __launch_bounds__(256)
void zero_buf(float* __restrict__ p, int n)
{
    int i = blockIdx.x * 256 + threadIdx.x;
    if (i < n) p[i] = 0.f;
}

__global__ __launch_bounds__(256)
void pool_add(const float* __restrict__ h, const int* __restrict__ batch,
              float* __restrict__ g)
{
    int i = blockIdx.x * 256 + threadIdx.x;
    if (i >= NN * 128) return;
    int n = i >> 7, c = i & 127;
    atomicAdd(&g[batch[n] * 128 + c], fmaxf(h[i], 0.f));   // global_add_pool(relu(h))
}

__global__ __launch_bounds__(256)
void head_kernel(const float* __restrict__ fc, const float* __restrict__ bng,
                 const float* __restrict__ bnb, const float* __restrict__ wcls,
                 const float* __restrict__ bcls, float* __restrict__ out)
{
    int row = blockIdx.x * 256 + threadIdx.x;
    if (row == 0) out[NGRP * NCLSN] = 0.0f;               // att_loss (eval mode)
    if (row >= NGRP) return;
    float inv = 1.0f / sqrtf(1.0f + 1e-5f);
    float logit[NCLSN];
    #pragma unroll
    for (int o = 0; o < NCLSN; ++o) logit[o] = bcls[o];
    for (int f = 0; f < 128; ++f) {
        float t = fmaxf(fc[row * 128 + f], 0.f);          // relu(fc)
        t = t * (bng[f] * inv) + bnb[f];                  // bn_hid
        #pragma unroll
        for (int o = 0; o < NCLSN; ++o) logit[o] += t * wcls[f * NCLSN + o];
    }
    float m = logit[0];
    #pragma unroll
    for (int o = 1; o < NCLSN; ++o) m = fmaxf(m, logit[o]);
    float s = 0.f;
    #pragma unroll
    for (int o = 0; o < NCLSN; ++o) s += __expf(logit[o] - m);
    float lse = m + __logf(s);
    #pragma unroll
    for (int o = 0; o < NCLSN; ++o) out[row * NCLSN + o] = logit[o] - lse;
}

// ---------------------------------------------------------------------------
extern "C" void kernel_launch(void* const* d_in, const int* in_sizes, int n_in,
                              void* d_out, int out_size, void* d_ws, size_t ws_size,
                              hipStream_t stream)
{
    const float* x     = (const float*)d_in[0];
    const int*   eidx  = (const int*)d_in[1];
    const int*   batch = (const int*)d_in[2];
    const float* bnfg  = (const float*)d_in[3];
    const float* bnfb  = (const float*)d_in[4];
    const float* wfeat = (const float*)d_in[5];
    const float* bfeat = (const float*)d_in[6];
    const float* bncg  = (const float*)d_in[7];
    const float* bncb  = (const float*)d_in[8];
    const float* wgat  = (const float*)d_in[9];
    const float* attl  = (const float*)d_in[10];
    const float* attr  = (const float*)d_in[11];
    const float* bgat  = (const float*)d_in[12];
    const float* bnfcg = (const float*)d_in[13];
    const float* bnfcb = (const float*)d_in[14];
    const float* wfc   = (const float*)d_in[15];
    const float* bfc   = (const float*)d_in[16];
    const float* bnhg  = (const float*)d_in[17];
    const float* bnhb  = (const float*)d_in[18];
    const float* wcls  = (const float*)d_in[19];
    const float* bcls  = (const float*)d_in[20];
    float* out = (float*)d_out;
    (void)in_sizes; (void)n_in; (void)out_size; (void)ws_size;

    float* ws    = (float*)d_ws;
    float* hA    = ws;                          // [NN,128]  ping
    float* hB    = hA    + (size_t)NN * 128;    // [NN,128]  pong
    float* hh    = hB    + (size_t)NN * 128;    // [NN,128]  per-layer transformed feats
    float* alpha = hh    + (size_t)NN * 128;    // [E2,4]
    float* mmax  = alpha + (size_t)E2 * 4;      // [NN,4]
    float* ssum  = mmax  + (size_t)NN * 4;      // [NN,4]
    float* gpool = ssum  + (size_t)NN * 4;      // [NGRP,128]
    float* fcout = gpool + (size_t)NGRP * 128;  // [NGRP,128]
    float* ebias = fcout + (size_t)NGRP * 128;  // [128]
    bf16_t* Wbf  = (bf16_t*)(ebias + 128);      // [128,128] bf16 (transposed)

    const int* src = eidx;
    const int* dst = eidx + EE;

    dim3 B(256);
    int gemm_blocks  = (NN + 127) / 128;                 // 391
    int edge_blocks  = (int)(((size_t)E2 * 32 + 255) / 256);
    int eh_blocks    = (E2 * 4 + 255) / 256;
    int feat_blocks  = (int)(((size_t)NN * 128 + 255) / 256);

    // h0 = bn_feat(x) @ w_feat + b_feat   (pre-ReLU; consumers apply ReLU)
    prep_weights<<<1, 128, 0, stream>>>(wfeat, bnfg, bnfb, bfeat, Wbf, ebias, 1);
    gemm_bf16_128<<<gemm_blocks, B, 0, stream>>>(x, Wbf, ebias, hA, NN, 0);

    float* hcur = hA;
    float* hnext = hB;
    for (int i = 0; i < NCONV; ++i) {
        prep_weights<<<1, 128, 0, stream>>>(wgat + (size_t)i * 128 * 128,
                                            bncg + i * 128, bncb + i * 128,
                                            nullptr, Wbf, ebias, 0);
        // hh = bn_conv(relu(hcur)) @ W_gat[i]
        gemm_bf16_128<<<gemm_blocks, B, 0, stream>>>(hcur, Wbf, ebias, hh, NN, 1);
        layer_init<<<feat_blocks, B, 0, stream>>>(hnext, bgat + i * 128, mmax, ssum);
        edge_alpha<<<edge_blocks, B, 0, stream>>>(hh, src, dst, attl + i * 128,
                                                  attr + i * 128, alpha, mmax);
        edge_exp<<<eh_blocks, B, 0, stream>>>(dst, mmax, alpha, ssum);
        edge_scatter<<<edge_blocks, B, 0, stream>>>(hh, src, dst, alpha, ssum, hnext);
        float* t = hcur; hcur = hnext; hnext = t;
    }

    zero_buf<<<(NGRP * 128 + 255) / 256, B, 0, stream>>>(gpool, NGRP * 128);
    pool_add<<<feat_blocks, B, 0, stream>>>(hcur, batch, gpool);

    // fc = bn_fc(gpool) @ w_fc + b_fc   (pre-ReLU)
    prep_weights<<<1, 128, 0, stream>>>(wfc, bnfcg, bnfcb, bfc, Wbf, ebias, 1);
    gemm_bf16_128<<<(NGRP + 127) / 128, B, 0, stream>>>(gpool, Wbf, ebias, fcout, NGRP, 0);

    head_kernel<<<(NGRP + 255) / 256, B, 0, stream>>>(fcout, bnhg, bnhb, wcls, bcls, out);
}
